// Conv2dTB_51668456571366
// MI455X (gfx1250) — matrix operand order repaired and checked by reading, not compiled, not run on hardware
//
#include <hip/hip_runtime.h>
#include <hip/hip_bf16.h>

// ---------------------------------------------------------------------------
// Problem constants (from reference): N=32, Cin=Cout=64, H=W=112, 3x3 pad=1
// ---------------------------------------------------------------------------
#define NB   32
#define CH   64
#define HH   112
#define WW   112
#define HW   (HH * WW)              // 12544
#define NPIX (NB * HW)              // 401408
#define KTOT 576                    // 9 * 64, k = r*64 + cin
#define THRESH 0.6f
#define BNEPS  1e-4f

typedef __attribute__((ext_vector_type(16))) _Float16 v16h;
typedef __attribute__((ext_vector_type(8)))  _Float16 v8h;
typedef __attribute__((ext_vector_type(8)))  float    v8f;

// ---------------------------------------------------------------------------
// Kernel 1: BatchNorm batch statistics -> per-channel scale/shift
// one block per channel, 256 threads reduce 401408 elements
// ---------------------------------------------------------------------------
__global__ __launch_bounds__(256) void bnstats_kernel(
    const float* __restrict__ x, const float* __restrict__ gamma,
    const float* __restrict__ bnbeta,
    float* __restrict__ scale, float* __restrict__ shift)
{
    const int c = blockIdx.x;
    const int tid = threadIdx.x;
    float s = 0.f, s2 = 0.f;
    for (int n = 0; n < NB; ++n) {
        const float* xp = x + ((size_t)n * CH + c) * HW;
        for (int i = tid; i < HW; i += 256) {
            float v = xp[i];
            s += v;
            s2 += v * v;
        }
    }
    __shared__ float rs[256], rq[256];
    rs[tid] = s; rq[tid] = s2;
    __syncthreads();
    for (int off = 128; off > 0; off >>= 1) {
        if (tid < off) { rs[tid] += rs[tid + off]; rq[tid] += rq[tid + off]; }
        __syncthreads();
    }
    if (tid == 0) {
        const float inv = 1.0f / (float)NPIX;
        float mean = rs[0] * inv;
        float var  = rq[0] * inv - mean * mean;       // biased variance (jnp.var)
        float sc   = gamma[c] * rsqrtf(var + BNEPS);
        scale[c] = sc;
        shift[c] = bnbeta[c] - mean * sc;
    }
}

// ---------------------------------------------------------------------------
// Kernel 2: ternary quantization. One thread per pixel (n,h,w):
//   - reads x NCHW (coalesced across threads per channel iteration)
//   - writes t in NHWC f16 (channel-contiguous for implicit-GEMM A fetch)
//   - writes per-pixel c_sum / c_num
// ---------------------------------------------------------------------------
__global__ __launch_bounds__(256) void ternary_kernel(
    const float* __restrict__ x, const float* __restrict__ scale,
    const float* __restrict__ shift, _Float16* __restrict__ t,
    float* __restrict__ csum, float* __restrict__ cnum)
{
    __shared__ float ss[CH], sh[CH];
    const int tid = threadIdx.x;
    if (tid < CH) { ss[tid] = scale[tid]; sh[tid] = shift[tid]; }
    __syncthreads();

    const int p  = blockIdx.x * 256 + tid;     // < 401408 exactly
    const int n  = p / HW;
    const int hw = p % HW;
    const float* xp = x + (size_t)n * CH * HW + hw;

    float cs = 0.f, cn = 0.f;
    _Float16 q[CH];
#pragma unroll
    for (int c = 0; c < CH; ++c) {
        float v  = xp[(size_t)c * HW];
        float xn = v * ss[c] + sh[c];
        float xc = fminf(1.f, fmaxf(-1.f, xn));
        float o  = (xc >= THRESH) ? 1.f : ((xc <= -THRESH) ? -1.f : 0.f);
        cs += xc * o;                 // == |xc| where quantized nonzero
        cn += fabsf(o);
        q[c] = (_Float16)o;
    }
    v8h* tp = (v8h*)(t + (size_t)p * CH);
#pragma unroll
    for (int i = 0; i < 8; ++i) tp[i] = *(const v8h*)(q + i * 8);
    csum[p] = cs;
    cnum[p] = cn;
}

// ---------------------------------------------------------------------------
// Kernel 3: beta = (box3x3(c_sum)+bias) / fixup(box3x3(c_num)+bias)
// ---------------------------------------------------------------------------
__global__ __launch_bounds__(256) void beta_kernel(
    const float* __restrict__ csum, const float* __restrict__ cnum,
    const float* __restrict__ beta_bias, float* __restrict__ beta)
{
    const int p  = blockIdx.x * 256 + threadIdx.x;
    const int n  = p / HW;
    const int hw = p % HW;
    const int h  = hw / WW;
    const int w  = hw % WW;
    const float* cs = csum + (size_t)n * HW;
    const float* cn = cnum + (size_t)n * HW;
    float ssum = 0.f, snum = 0.f;
#pragma unroll
    for (int dy = -1; dy <= 1; ++dy) {
#pragma unroll
        for (int dx = -1; dx <= 1; ++dx) {
            int hh = h + dy, ww = w + dx;
            if (hh >= 0 && hh < HH && ww >= 0 && ww < WW) {
                int o = hh * WW + ww;
                ssum += cs[o];
                snum += cn[o];
            }
        }
    }
    float bb  = beta_bias[0];
    float cnt = snum + bb;
    if (cnt == 0.f) cnt = 1.f;
    beta[p] = (ssum + bb) / cnt;
}

// ---------------------------------------------------------------------------
// Kernel 4: weight repack fp32 [Cout][Cin][3][3] -> f16 [Cout][k], k = r*64+cin
// ---------------------------------------------------------------------------
__global__ __launch_bounds__(256) void wprep_kernel(
    const float* __restrict__ w, _Float16* __restrict__ wt)
{
    const int i = blockIdx.x * 256 + threadIdx.x;   // < 64*576 = 36864
    const int cout = i / KTOT;
    const int k    = i % KTOT;
    const int r    = k / CH;
    const int cin  = k % CH;
    wt[i] = (_Float16)w[((size_t)cout * CH + cin) * 9 + r];
}

// ---------------------------------------------------------------------------
// Kernel 5: implicit-GEMM 3x3 conv via v_wmma_f32_16x16x32_f16.
//   block = 256 threads = 8 waves; wave = 16 output pixels (one row segment)
//   x 64 couts (4 v8f accumulators). K loop: 18 steps of K=32
//   (filter tap r fixed per step, 32 contiguous input channels).
//   B tile (64x32 f16) double-buffered through LDS via CDNA5 async
//   load-to-LDS (global_load_async_to_lds_b128 + s_wait_asynccnt), so the
//   weight prefetch of step s+1 overlaps the 4 WMMAs of step s.
// ---------------------------------------------------------------------------
__global__ __launch_bounds__(256) void conv_wmma_kernel(
    const _Float16* __restrict__ t,     // NHWC  [N][H][W][64]
    const _Float16* __restrict__ wt,    // [64][576] f16
    const float* __restrict__ convb,    // [64]
    const float* __restrict__ beta,     // [N*H*W]
    float* __restrict__ out)            // NCHW  [N][64][H][W]
{
    __shared__ _Float16 sB[2][CH][32];      // 8 KB, double-buffered weight tile
    __shared__ float    sOut[8][16][17];    // per-wave epilogue transpose

    const int tid  = threadIdx.x;
    const int wid  = tid >> 5;
    const int lane = tid & 31;

    // tile id: N * 112 rows * 7 sixteen-pixel segments = 25088, /8 waves = 3136 blocks
    const int tile = blockIdx.x * 8 + wid;
    const int n    = tile / (HH * 7);
    const int rem  = tile % (HH * 7);
    const int oh   = rem / 7;
    const int ow0  = (rem % 7) * 16;

    const int mrow = lane & 15;             // A row (pixel within tile)
    const int kloA = (lane & 16) ? 8 : 0;   // A layout: low lanes K0-7/16-23, high K8-15/24-31
    const int kloB = (lane & 16) ? 8 : 0;   // B layout: low lanes K0-7/16-23, high K8-15/24-31

    // cooperative async B-tile fill: thread -> (cout = tid/4, 8 halfs @ (tid%4)*8)
    // Each lane issues one 16-byte async copy global -> LDS; data visibility is
    // established by s_wait_asynccnt 0 followed by the block barrier.
    const int fco = tid >> 2;
    const int fkq = (tid & 3) << 3;
    auto fillB = [&](int buf, int s) {
        const int r    = s >> 1;
        const int cin0 = (s & 1) << 5;
        const int K0   = r * CH + cin0;
        // low 32 bits of a generic pointer to __shared__ == LDS byte offset
        unsigned lds_addr = (unsigned)(size_t)(void*)&sB[buf][fco][fkq];
        unsigned long long gaddr =
            (unsigned long long)(size_t)(wt + (size_t)fco * KTOT + K0 + fkq);
        asm volatile("global_load_async_to_lds_b128 %0, %1, off"
                     :: "v"(lds_addr), "v"(gaddr) : "memory");
    };

    v8f acc[4];
#pragma unroll
    for (int nt = 0; nt < 4; ++nt) acc[nt] = (v8f){};

    fillB(0, 0);
    asm volatile("s_wait_asynccnt 0x0" ::: "memory");
    __syncthreads();

    const v8h zero8 = {};
    for (int s = 0; s < 18; ++s) {
        const int buf = s & 1;
        if (s + 1 < 18) fillB((s + 1) & 1, s + 1);   // async prefetch next tile

        // ---- A fragment: 16 halfs from NHWC t (zero-padded halo) ----
        const int r    = s >> 1;
        const int cin0 = (s & 1) << 5;
        const int dy   = r / 3 - 1;
        const int dx   = r % 3 - 1;
        const int ih   = oh + dy;
        const int iw   = ow0 + mrow + dx;
        const bool valid = (ih >= 0) & (ih < HH) & (iw >= 0) & (iw < WW);
        const _Float16* ap =
            t + (((size_t)n * HH + ih) * WW + iw) * CH + cin0;
        union { v16h v; v8h h[2]; } a;
        a.h[0] = valid ? *(const v8h*)(ap + kloA)      : zero8;
        a.h[1] = valid ? *(const v8h*)(ap + kloA + 16) : zero8;

        // ---- 4 WMMAs (cout tiles 0..3) from the LDS weight tile ----
#pragma unroll
        for (int nt = 0; nt < 4; ++nt) {
            union { v16h v; v8h h[2]; } b;
            const _Float16* bp = &sB[buf][nt * 16 + (lane & 15)][kloB];
            b.h[0] = *(const v8h*)(bp);
            b.h[1] = *(const v8h*)(bp + 16);
            acc[nt] = __builtin_amdgcn_wmma_f32_16x16x32_f16(
                false, a.v, false, b.v, (short)0, acc[nt], false, false);
        }
        // complete the async prefetch, then make it visible block-wide
        asm volatile("s_wait_asynccnt 0x0" ::: "memory");
        __syncthreads();
    }

    // ---- epilogue: transpose via LDS so NCHW stores are contiguous ----
    const int pix   = lane & 15;
    const float bb  = beta[((size_t)n * HH + oh) * WW + ow0 + pix];
    const int mbase = (lane & 16) ? 8 : 0;
#pragma unroll
    for (int nt = 0; nt < 4; ++nt) {
        __syncthreads();
#pragma unroll
        for (int g = 0; g < 8; ++g)
            sOut[wid][lane & 15][mbase + g] = acc[nt][g];   // [col=cout][row=pixel]
        __syncthreads();
#pragma unroll
        for (int j = 0; j < 8; ++j) {
            const int crow = j * 2 + (lane >> 4);           // cout within tile
            const int cout = nt * 16 + crow;
            float v = sOut[wid][crow][pix];
            out[(((size_t)n * CH + cout) * HH + oh) * WW + ow0 + pix] =
                (v + convb[cout]) * bb;
        }
    }
}

// ---------------------------------------------------------------------------
// Host launcher
// Workspace layout (bytes):
//   [0,256)        scale (64 f32)
//   [256,512)      shift (64 f32)
//   [512, +51380224)        t  (NHWC f16, 32*112*112*64)
//   then csum (401408 f32), cnum (401408 f32), beta (401408 f32),
//   then wt (64*576 f16)   -> total ~56.3 MB
// ---------------------------------------------------------------------------
extern "C" void kernel_launch(void* const* d_in, const int* in_sizes, int n_in,
                              void* d_out, int out_size, void* d_ws, size_t ws_size,
                              hipStream_t stream) {
    const float* x      = (const float*)d_in[0];
    const float* gamma  = (const float*)d_in[1];
    const float* bnbeta = (const float*)d_in[2];
    const float* convw  = (const float*)d_in[3];
    const float* convb  = (const float*)d_in[4];
    const float* betab  = (const float*)d_in[5];

    char* ws = (char*)d_ws;
    float*    scale = (float*)(ws + 0);
    float*    shift = (float*)(ws + 256);
    _Float16* t     = (_Float16*)(ws + 512);
    const size_t tbytes = (size_t)NPIX * CH * sizeof(_Float16);   // 51,380,224
    float*    csum  = (float*)(ws + 512 + tbytes);
    float*    cnum  = csum + NPIX;
    float*    beta  = cnum + NPIX;
    _Float16* wt    = (_Float16*)(beta + NPIX);

    bnstats_kernel<<<CH, 256, 0, stream>>>(x, gamma, bnbeta, scale, shift);
    ternary_kernel<<<NPIX / 256, 256, 0, stream>>>(x, scale, shift, t, csum, cnum);
    beta_kernel<<<NPIX / 256, 256, 0, stream>>>(csum, cnum, betab, beta);
    wprep_kernel<<<(CH * KTOT) / 256, 256, 0, stream>>>(convw, wt);
    conv_wmma_kernel<<<(NB * HH * 7) / 8, 256, 0, stream>>>(
        t, wt, convb, beta, (float*)d_out);
}
